// SelfAttention_v2_57518202028174
// MI455X (gfx1250) — compile-verified, matrix operand order checked by reading, not run on hardware
//
#include <hip/hip_runtime.h>

// ---------------------------------------------------------------------------
// Self-attention (N=8192, D=1024) for gfx1250 / MI455X.
//   1) convert x, Wq, Wk, Wv  f32 -> bf16 (workspace)
//   2) Q = x@Wq^T, K = x@Wk^T  (row-major bf16, f32 accum WMMA)
//      V^T = Wv @ x^T          (transposed output -> PV GEMM needs no LDS
//                               transpose; B-fragments are contiguous rows)
//   3) flash attention: 1024-thread WG owns 64 query rows, streams 128-key
//      blocks; online softmax in f32; S and P@V both on
//      v_wmma_f32_16x16x32_bf16. K (16MB) and V^T (16MB) live in the 192MB L2.
// Workspace: (8192*1024 + 3*1024*1024 + 3*8192*1024) * 2B ~= 70 MiB.
// ---------------------------------------------------------------------------

typedef __attribute__((ext_vector_type(16))) __bf16 bf16x16;
typedef __attribute__((ext_vector_type(8)))  float  f32x8;

#define AN 8192
#define AD 1024

#define WMMA_BF16(a, b, c) \
  __builtin_amdgcn_wmma_f32_16x16x32_bf16(false, (a), false, (b), (short)0, (c), false, false)

// float -> bf16 bits, round-to-nearest-even
static __device__ __forceinline__ unsigned short f2bf(float x) {
  unsigned int u = __float_as_uint(x);
  unsigned int r = 0x7FFFu + ((u >> 16) & 1u);
  return (unsigned short)((u + r) >> 16);
}

// A/B fragment from a row-major bf16 matrix in global memory.
// ISA 16-bit A layout: lane<16 -> row, K 0..7 & 16..23 ; lane>=16 -> same row, K 8..15 & 24..31.
static __device__ __forceinline__ bf16x16 frag_rows_g(const unsigned short* __restrict__ base,
                                                      int ld, int row0, int k0) {
  const int lane = threadIdx.x & 31;
  const int r    = lane & 15;
  const int koff = (lane >> 4) << 3;
  const uint4* p = (const uint4*)(base + (size_t)(row0 + r) * ld + (k0 + koff));
  union { struct { uint4 lo, hi; } q; bf16x16 v; } c;
  c.q.lo = p[0];   // K = k0+koff    .. +7
  c.q.hi = p[2];   // K = k0+koff+16 .. +23
  return c.v;
}

// Same fragment gather from an LDS row-major buffer (stride in u16, 16B-aligned rows).
static __device__ __forceinline__ bf16x16 frag_rows_lds(const unsigned short* base,
                                                        int ld, int row0, int k0) {
  const int lane = threadIdx.x & 31;
  const int r    = lane & 15;
  const int koff = (lane >> 4) << 3;
  const uint4* p = (const uint4*)(base + (row0 + r) * ld + (k0 + koff));
  union { struct { uint4 lo, hi; } q; bf16x16 v; } c;
  c.q.lo = p[0];
  c.q.hi = p[2];
  return c.v;
}

// ---------------------------------------------------------------------------
__global__ __launch_bounds__(512) void cvt_f32_to_bf16(const float* __restrict__ src,
                                                       unsigned short* __restrict__ dst, int n) {
  int i = blockIdx.x * blockDim.x + threadIdx.x;
  if (i < n) dst[i] = f2bf(src[i]);
}

// ---------------------------------------------------------------------------
// Q/K = xb @ W^T, row-major outputs. Wave tile 32x64 (8 C-tiles),
// block = 8 waves (2x4) -> 64x256, grid (128,4,2) (z: 0=Q, 1=K).
__global__ __launch_bounds__(256) void qk_gemm(
    const unsigned short* __restrict__ xb,
    const unsigned short* __restrict__ Wqb,
    const unsigned short* __restrict__ Wkb,
    unsigned short* __restrict__ Qb,
    unsigned short* __restrict__ Kb) {
  const unsigned short* W = (blockIdx.z == 0) ? Wqb : Wkb;
  unsigned short*       Y = (blockIdx.z == 0) ? Qb  : Kb;

  const int wave = threadIdx.x >> 5;
  const int wr   = wave >> 2;
  const int wcn  = wave & 3;
  const int row0 = blockIdx.x * 64 + wr * 32;
  const int col0 = blockIdx.y * 256 + wcn * 64;

  f32x8 acc[2][4] = {};
#pragma unroll 4
  for (int k = 0; k < AD; k += 32) {
    bf16x16 a0 = frag_rows_g(xb, AD, row0, k);
    bf16x16 a1 = frag_rows_g(xb, AD, row0 + 16, k);
    bf16x16 b0 = frag_rows_g(W, AD, col0 + 0,  k);
    bf16x16 b1 = frag_rows_g(W, AD, col0 + 16, k);
    bf16x16 b2 = frag_rows_g(W, AD, col0 + 32, k);
    bf16x16 b3 = frag_rows_g(W, AD, col0 + 48, k);
    acc[0][0] = WMMA_BF16(a0, b0, acc[0][0]);
    acc[0][1] = WMMA_BF16(a0, b1, acc[0][1]);
    acc[0][2] = WMMA_BF16(a0, b2, acc[0][2]);
    acc[0][3] = WMMA_BF16(a0, b3, acc[0][3]);
    acc[1][0] = WMMA_BF16(a1, b0, acc[1][0]);
    acc[1][1] = WMMA_BF16(a1, b1, acc[1][1]);
    acc[1][2] = WMMA_BF16(a1, b2, acc[1][2]);
    acc[1][3] = WMMA_BF16(a1, b3, acc[1][3]);
  }

  const int lane = threadIdx.x & 31;
  const int cr = lane & 15;
  const int rp = (lane >> 4) << 3;
#pragma unroll
  for (int mi = 0; mi < 2; ++mi)
#pragma unroll
    for (int j = 0; j < 4; ++j)
#pragma unroll
      for (int r = 0; r < 8; ++r)
        Y[(size_t)(row0 + mi * 16 + r + rp) * AD + (col0 + 16 * j + cr)] = f2bf(acc[mi][j][r]);
}

// ---------------------------------------------------------------------------
// V^T[c][n] = sum_k Wv[c][k] * x[n][k]  -> A = Wv rows, B lanes = x rows.
// Output VtG is [1024][8192] row-major (stores coalesced: lanes run along n).
__global__ __launch_bounds__(256) void v_gemm_transposed(
    const unsigned short* __restrict__ xb,
    const unsigned short* __restrict__ Wvb,
    unsigned short* __restrict__ VtG) {
  const int wave = threadIdx.x >> 5;
  const int wr   = wave >> 2;
  const int wcn  = wave & 3;
  const int row0 = blockIdx.x * 64 + wr * 32;    // c dim (0..1023)
  const int col0 = blockIdx.y * 256 + wcn * 64;  // n dim (0..8191)

  f32x8 acc[2][4] = {};
#pragma unroll 4
  for (int k = 0; k < AD; k += 32) {
    bf16x16 a0 = frag_rows_g(Wvb, AD, row0, k);
    bf16x16 a1 = frag_rows_g(Wvb, AD, row0 + 16, k);
    bf16x16 b0 = frag_rows_g(xb, AD, col0 + 0,  k);
    bf16x16 b1 = frag_rows_g(xb, AD, col0 + 16, k);
    bf16x16 b2 = frag_rows_g(xb, AD, col0 + 32, k);
    bf16x16 b3 = frag_rows_g(xb, AD, col0 + 48, k);
    acc[0][0] = WMMA_BF16(a0, b0, acc[0][0]);
    acc[0][1] = WMMA_BF16(a0, b1, acc[0][1]);
    acc[0][2] = WMMA_BF16(a0, b2, acc[0][2]);
    acc[0][3] = WMMA_BF16(a0, b3, acc[0][3]);
    acc[1][0] = WMMA_BF16(a1, b0, acc[1][0]);
    acc[1][1] = WMMA_BF16(a1, b1, acc[1][1]);
    acc[1][2] = WMMA_BF16(a1, b2, acc[1][2]);
    acc[1][3] = WMMA_BF16(a1, b3, acc[1][3]);
  }

  const int lane = threadIdx.x & 31;
  const int cr = lane & 15;
  const int rp = (lane >> 4) << 3;
#pragma unroll
  for (int mi = 0; mi < 2; ++mi)
#pragma unroll
    for (int j = 0; j < 4; ++j)
#pragma unroll
      for (int r = 0; r < 8; ++r)
        VtG[(size_t)(row0 + mi * 16 + r + rp) * AN + (col0 + 16 * j + cr)] = f2bf(acc[mi][j][r]);
}

// ---------------------------------------------------------------------------
// Flash attention. Block = 1024 threads = 32 waves (4 row-groups x 8 col-waves),
// owns 64 query rows. Key blocks of 128. Each wave owns O[16 x 128].
// P@V reads V^T directly from global (L2-resident) -> no LDS transpose.
__global__ __launch_bounds__(1024) void flash_attn_kernel(
    const unsigned short* __restrict__ Qb,
    const unsigned short* __restrict__ Kb,
    const unsigned short* __restrict__ VtG,
    float* __restrict__ out) {
  __shared__ __align__(16) unsigned short Pbuf[64 * 136];  // P bf16: [qrow][key0..127], padded stride
  __shared__ float redm[64][8];
  __shared__ float redsum[64][8];
  __shared__ float rowM[64];
  __shared__ float rowAlpha[64];
  __shared__ float rowL[64];

  const float scale = 0.03125f;                 // 1/sqrt(1024)
  const float LOG2E = 1.44269504088896340736f;

  const int tid    = threadIdx.x;
  const int lane   = tid & 31;
  const int wave   = tid >> 5;       // 0..31
  const int rg     = wave >> 3;      // row group 0..3
  const int wc     = wave & 7;       // key/col group 0..7
  const int qrow0  = blockIdx.x * 64;
  const int myrow0 = qrow0 + rg * 16;
  const int cr = lane & 15;
  const int rp = (lane >> 4) << 3;

  if (tid < 64) { rowM[tid] = -__builtin_inff(); rowL[tid] = 0.0f; }
  __syncthreads();

  f32x8 O[8] = {};   // 16 rows x 128 cols, col tile t at wc*128 + 16*t

  for (int kb = 0; kb < AN; kb += 128) {
    const int key0 = kb + wc * 16;

    // kick next key block's K rows toward the WGP caches (global_prefetch_b8)
    if (kb + 128 < AN)
      __builtin_prefetch(Kb + (size_t)(key0 + 128 + cr) * AD, 0, 1);

    // ---- S tile: rows myrow0..+15, keys key0..+15 (f32 accum) -------------
    f32x8 s = {};
#pragma unroll 8
    for (int k = 0; k < AD; k += 32) {
      bf16x16 a = frag_rows_g(Qb, AD, myrow0, k);
      bf16x16 b = frag_rows_g(Kb, AD, key0, k);
      s = WMMA_BF16(a, b, s);
    }
#pragma unroll
    for (int r = 0; r < 8; ++r) s[r] *= scale;

    // ---- per-wave row max (butterfly within 16-lane halves) ---------------
#pragma unroll
    for (int r = 0; r < 8; ++r) {
      float m = s[r];
      m = fmaxf(m, __shfl_xor(m, 1, 32));
      m = fmaxf(m, __shfl_xor(m, 2, 32));
      m = fmaxf(m, __shfl_xor(m, 4, 32));
      m = fmaxf(m, __shfl_xor(m, 8, 32));
      if (cr == 0) redm[rg * 16 + r + rp][wc] = m;
    }
    __syncthreads();

    // ---- combine maxima, compute rescale factor ---------------------------
    if (tid < 64) {
      float mb = redm[tid][0];
#pragma unroll
      for (int c = 1; c < 8; ++c) mb = fmaxf(mb, redm[tid][c]);
      float mo = rowM[tid];
      float mn = fmaxf(mo, mb);
      rowM[tid] = mn;
      rowAlpha[tid] = exp2f((mo - mn) * LOG2E);   // first block: exp2(-inf)=0
    }
    __syncthreads();

    // ---- P = exp(s - m), row sums, stage P (bf16), rescale O --------------
#pragma unroll
    for (int r = 0; r < 8; ++r) {
      const int row = rg * 16 + r + rp;
      float p = exp2f((s[r] - rowM[row]) * LOG2E);
      float sum = p;
      sum += __shfl_xor(sum, 1, 32);
      sum += __shfl_xor(sum, 2, 32);
      sum += __shfl_xor(sum, 4, 32);
      sum += __shfl_xor(sum, 8, 32);
      if (cr == 0) redsum[row][wc] = sum;
      Pbuf[row * 136 + wc * 16 + cr] = f2bf(p);
    }
#pragma unroll
    for (int r = 0; r < 8; ++r) {
      const float al = rowAlpha[rg * 16 + r + rp];
#pragma unroll
      for (int t = 0; t < 8; ++t) O[t][r] *= al;
    }
    __syncthreads();

    if (tid < 64) {
      float sb = 0.0f;
#pragma unroll
      for (int c = 0; c < 8; ++c) sb += redsum[tid][c];
      rowL[tid] = rowL[tid] * rowAlpha[tid] + sb;
    }

    // ---- O += P(16x128) @ V(128x1024) via V^T rows (pure WMMA + b128) -----
#pragma unroll
    for (int kk2 = 0; kk2 < 128; kk2 += 32) {
      bf16x16 a = frag_rows_lds(Pbuf, 136, rg * 16, kk2);
#pragma unroll
      for (int t = 0; t < 8; ++t) {
        bf16x16 b = frag_rows_g(VtG, AN, wc * 128 + t * 16, kb + kk2);
        O[t] = WMMA_BF16(a, b, O[t]);
      }
    }
    // Pbuf is next overwritten only after two more barriers -> no sync needed here
  }

  // ---- normalize and store f32 output -------------------------------------
  __syncthreads();
#pragma unroll
  for (int r = 0; r < 8; ++r) {
    const int row = rg * 16 + r + rp;
    const float inv = 1.0f / rowL[row];
#pragma unroll
    for (int t = 0; t < 8; ++t)
      out[(size_t)(qrow0 + row) * AD + (wc * 128 + t * 16 + cr)] = O[t][r] * inv;
  }
}

// ---------------------------------------------------------------------------
extern "C" void kernel_launch(void* const* d_in, const int* in_sizes, int n_in,
                              void* d_out, int out_size, void* d_ws, size_t ws_size,
                              hipStream_t stream) {
  (void)in_sizes; (void)n_in; (void)out_size; (void)ws_size;
  const float* x  = (const float*)d_in[0];
  const float* Wq = (const float*)d_in[1];
  const float* Wk = (const float*)d_in[2];
  const float* Wv = (const float*)d_in[3];
  float* out = (float*)d_out;

  const size_t NX = (size_t)AN * AD;   // 8388608
  const size_t NW = (size_t)AD * AD;   // 1048576

  unsigned short* ws  = (unsigned short*)d_ws;
  unsigned short* xb  = ws;
  unsigned short* Wqb = xb + NX;
  unsigned short* Wkb = Wqb + NW;
  unsigned short* Wvb = Wkb + NW;
  unsigned short* Qb  = Wvb + NW;
  unsigned short* Kb  = Qb + NX;
  unsigned short* VtG = Kb + NX;       // [1024][8192] bf16

  cvt_f32_to_bf16<<<(unsigned)((NX + 511) / 512), 512, 0, stream>>>(x,  xb,  (int)NX);
  cvt_f32_to_bf16<<<(unsigned)((NW + 511) / 512), 512, 0, stream>>>(Wq, Wqb, (int)NW);
  cvt_f32_to_bf16<<<(unsigned)((NW + 511) / 512), 512, 0, stream>>>(Wk, Wkb, (int)NW);
  cvt_f32_to_bf16<<<(unsigned)((NW + 511) / 512), 512, 0, stream>>>(Wv, Wvb, (int)NW);

  qk_gemm<<<dim3(AN / 64, AD / 256, 2), 256, 0, stream>>>(xb, Wqb, Wkb, Qb, Kb);
  v_gemm_transposed<<<dim3(AD / 64, AN / 256), 256, 0, stream>>>(xb, Wvb, VtG);

  flash_attn_kernel<<<dim3(AN / 64), 1024, 0, stream>>>(Qb, Kb, VtG, out);
}
